// Model_14499809591701
// MI455X (gfx1250) — matrix operand order repaired and checked by reading, not compiled, not run on hardware
//
#include <hip/hip_runtime.h>
#include <hip/hip_bf16.h>

typedef __attribute__((ext_vector_type(16))) __bf16 v16bf;
typedef __attribute__((ext_vector_type(8)))  __bf16 v8bf;
typedef __attribute__((ext_vector_type(8)))  float  v8f;
typedef __attribute__((ext_vector_type(4)))  float  v4f;

constexpr int kN  = 8192;   // nodes
constexpr int kDH = 64;     // hidden dim after encoder

__device__ __forceinline__ float leaky(float v) { return fmaxf(v, 0.1f * v); }

// ---------------------------------------------------------------------------
// Kernel 1: MLP encoder.  h = leaky(leaky(x@W1_0.T+b1_0)@W1_1.T+b1_1) [N,64]
// Also produces: ht (bf16, transposed [64][N] for WMMA B-fragments),
//                s = h@a, sq = ||h||^2/64.
// One block = 16 rows, 256 threads.
// ---------------------------------------------------------------------------
__global__ __launch_bounds__(256) void mlp_kernel(
    const float* __restrict__ x,
    const float* __restrict__ W10, const float* __restrict__ b10,
    const float* __restrict__ W11, const float* __restrict__ b11,
    const float* __restrict__ a,
    float* __restrict__ hf32, __bf16* __restrict__ ht,
    float* __restrict__ s_ws, float* __restrict__ sq_ws)
{
    const int t  = threadIdx.x;
    const int i0 = blockIdx.x * 16;

    __shared__ float xt[16 * 260];   // 256 + pad (bank-conflict-free, 16B aligned)
    __shared__ float h1[16 * 132];   // 128 + pad
    __shared__ float hl[16 * 68];    // 64 + pad
    __shared__ float red[2 * 16 * 16];

    // load x tile [16][256], coalesced
    for (int q = 0; q < 16; ++q)
        xt[q * 260 + t] = x[(size_t)(i0 + q) * 256 + t];
    __syncthreads();

    const int r = t & 15;     // row in tile
    const int u = t >> 4;     // 0..15

    // layer 1: 128 outputs, 8 per thread (o = u + 16p). W row broadcast per 16 lanes.
    for (int p = 0; p < 8; ++p) {
        const int o = u + 16 * p;
        float acc = b10[o];
        const float* wrow = W10 + o * 256;
        for (int k = 0; k < 256; k += 4) {
            v4f w  = *(const v4f*)(wrow + k);
            v4f xv = *(const v4f*)(&xt[r * 260 + k]);
            acc += w.x * xv.x + w.y * xv.y + w.z * xv.z + w.w * xv.w;
        }
        h1[r * 132 + o] = leaky(acc);
    }
    __syncthreads();

    // layer 2: 64 outputs, 4 per thread; fold in s and sq partials
    float ps = 0.f, psq = 0.f;
    for (int p = 0; p < 4; ++p) {
        const int o = u + 16 * p;
        float acc = b11[o];
        const float* wrow = W11 + o * 128;
        for (int k = 0; k < 128; k += 4) {
            v4f w  = *(const v4f*)(wrow + k);
            v4f hv = *(const v4f*)(&h1[r * 132 + k]);
            acc += w.x * hv.x + w.y * hv.y + w.z * hv.z + w.w * hv.w;
        }
        float hv = leaky(acc);
        hl[r * 68 + o] = hv;
        hf32[(size_t)(i0 + r) * kDH + o] = hv;
        ps  += hv * a[o];
        psq += hv * hv;
    }
    red[0 * 256 + r * 16 + u] = ps;
    red[1 * 256 + r * 16 + u] = psq;
    __syncthreads();

    if (t < 16) {
        float ssum = 0.f, qsum = 0.f;
        for (int c = 0; c < 16; ++c) {
            ssum += red[t * 16 + c];
            qsum += red[256 + t * 16 + c];
        }
        s_ws[i0 + t]  = ssum;
        sq_ws[i0 + t] = qsum * (1.0f / (float)kDH);
    }
    // transposed bf16 store: ht[o][i0..i0+15] (32B contiguous per thread)
    if (t < kDH) {
        const int o = t;
        unsigned int* dst = (unsigned int*)(ht + (size_t)o * kN + i0);
        for (int q = 0; q < 8; ++q) {
            __bf16 lo = (__bf16)hl[(2 * q) * 68 + o];
            __bf16 hi = (__bf16)hl[(2 * q + 1) * 68 + o];
            unsigned int pk = ((unsigned int)(*(unsigned short*)&hi) << 16)
                            |  (unsigned int)(*(unsigned short*)&lo);
            dst[q] = pk;
        }
    }
}

// ---------------------------------------------------------------------------
// Kernel 2: fused attention pass over A.  One block = 16 rows of S.
// Phase 1: E = exp(leaky(s_i - s_j))*A  -> f32 stats in regs, bf16 tile in LDS
// Phase 2: T += E @ x   via v_wmma_f32_16x16x32_bf16 (8 waves: 4 K-chunks x 2 N-groups)
// Epilogue: normalize, L1/L2 partials, GNN layer + output head.
// ---------------------------------------------------------------------------
__global__ __launch_bounds__(256) void spass_kernel(
    const float* __restrict__ A,
    const float* __restrict__ s_ws, const float* __restrict__ sq_ws,
    const __bf16* __restrict__ ht, const float* __restrict__ hf32,
    const float* __restrict__ Wg, const float* __restrict__ bg,
    const float* __restrict__ W2, const float* __restrict__ b2,
    float* __restrict__ out, float* __restrict__ l1part, float* __restrict__ l2part)
{
    const int t    = threadIdx.x;
    const int lane = t & 31;
    const int w    = t >> 5;       // wave id 0..7
    const int i0   = blockIdx.x * 16;

    constexpr int EP = 136;        // padded bf16 row stride for E tile
    __shared__ __bf16 E_lds[16 * EP];      // 16x128 bf16 E tile
    __shared__ float  Tst[4 * 16 * 64];    // per-kchunk partial T (reused as g_in)
    __shared__ float  Tfull[16 * 64];
    __shared__ float  gl[16 * 64];
    __shared__ float  sred[3 * 16 * 16];
    __shared__ float  rowinv[16];
    __shared__ float  rowstat[32];
    __shared__ float  s_si[16], s_sqi[16];

    if (t < 16) { s_si[t] = s_ws[i0 + t]; s_sqi[t] = sq_ws[i0 + t]; }
    __syncthreads();

    // phase-1 role: (row, 8-col chunk)
    const int pr = t >> 4;
    const int pc = t & 15;
    const float si = s_si[pr];
    float rs = 0.f, qs = 0.f, us = 0.f;

    // WMMA role: kchunk (32 K each) x N-group (2 x 16 cols)
    const int kc    = w & 3;
    const int gsel  = w >> 2;
    const int ml    = lane & 15;
    const int khalf = (lane < 16) ? 0 : 8;    // A-frag K offset (ISA 16-bit A layout)
    const int bhalf = (lane < 16) ? 0 : 8;    // B-frag K offset (same table layout as A)
    const int n0 = (2 * gsel + 0) * 16 + ml;
    const int n1 = (2 * gsel + 1) * 16 + ml;
    v8f c0 = {}; v8f c1 = {};

    const float* Arow = A + (size_t)(i0 + pr) * kN;

    for (int jb = 0; jb < kN; jb += 128) {
        // ---- phase 1: build E tile, accumulate row stats ----
        {
            const int j = jb + pc * 8;
            __builtin_prefetch(Arow + j + 128, 0, 0);
            v4f a0  = *(const v4f*)(Arow + j);
            v4f a1  = *(const v4f*)(Arow + j + 4);
            v4f sj0 = *(const v4f*)(s_ws + j);
            v4f sj1 = *(const v4f*)(s_ws + j + 4);
            v4f q0  = *(const v4f*)(sq_ws + j);
            v4f q1  = *(const v4f*)(sq_ws + j + 4);
            float Av[8] = {a0.x, a0.y, a0.z, a0.w, a1.x, a1.y, a1.z, a1.w};
            float Sv[8] = {sj0.x, sj0.y, sj0.z, sj0.w, sj1.x, sj1.y, sj1.z, sj1.w};
            float Qv[8] = {q0.x, q0.y, q0.z, q0.w, q1.x, q1.y, q1.z, q1.w};
            v8bf epk;
#pragma unroll
            for (int e = 0; e < 8; ++e) {
                float E = __expf(leaky(si - Sv[e])) * Av[e];
                rs += E; qs += E * E; us += E * Qv[e];
                epk[e] = (__bf16)E;
            }
            *(v8bf*)&E_lds[pr * EP + pc * 8] = epk;
        }
        __syncthreads();
        // ---- phase 2: WMMA accumulate T ----
        {
            union { v16bf v; v8bf h[2]; } af, bf0, bf1;
            const int kb = kc * 32 + khalf;
            af.h[0] = *(const v8bf*)&E_lds[ml * EP + kb];
            af.h[1] = *(const v8bf*)&E_lds[ml * EP + kb + 16];
            const int jcol = jb + kc * 32 + bhalf;
            const __bf16* bp0 = ht + (size_t)n0 * kN + jcol;
            const __bf16* bp1 = ht + (size_t)n1 * kN + jcol;
            bf0.h[0] = *(const v8bf*)(bp0);
            bf0.h[1] = *(const v8bf*)(bp0 + 16);
            bf1.h[0] = *(const v8bf*)(bp1);
            bf1.h[1] = *(const v8bf*)(bp1 + 16);
            c0 = __builtin_amdgcn_wmma_f32_16x16x32_bf16(false, af.v, false, bf0.v,
                                                         (short)0, c0, false, false);
            c1 = __builtin_amdgcn_wmma_f32_16x16x32_bf16(false, af.v, false, bf1.v,
                                                         (short)0, c1, false, false);
        }
        __syncthreads();
    }

    // ---- stage scalar partials and C accumulators ----
    sred[0 * 256 + pr * 16 + pc] = rs;
    sred[1 * 256 + pr * 16 + pc] = qs;
    sred[2 * 256 + pr * 16 + pc] = us;
    {
        const int Mofs = (lane < 16) ? 0 : 8;   // C layout: V v -> M=v (+8 upper lanes)
#pragma unroll
        for (int v = 0; v < 8; ++v) {
            Tst[kc * 1024 + (v + Mofs) * 64 + (2 * gsel + 0) * 16 + ml] = c0[v];
            Tst[kc * 1024 + (v + Mofs) * 64 + (2 * gsel + 1) * 16 + ml] = c1[v];
        }
    }
    __syncthreads();

    // reduce over kchunks -> Tfull[16][64]
    for (int q = 0; q < 4; ++q) {
        const int idx = t + 256 * q;
        Tfull[idx] = Tst[idx] + Tst[1024 + idx] + Tst[2048 + idx] + Tst[3072 + idx];
    }
    __syncthreads();

    // per-row stats: rowsum, Q, U, L1/L2 rows
    if (t < 16) {
        float rsum = 0.f, qsum = 0.f, usum = 0.f;
        for (int c = 0; c < 16; ++c) {
            rsum += sred[t * 16 + c];
            qsum += sred[256 + t * 16 + c];
            usum += sred[512 + t * 16 + c];
        }
        const float inv = 1.0f / rsum;
        float dot = 0.f;
        const float* hrow = hf32 + (size_t)(i0 + t) * kDH;
        for (int d = 0; d < kDH; ++d) dot += hrow[d] * Tfull[t * 64 + d];
        const float P = s_sqi[t] * rsum + usum - (2.0f / (float)kDH) * dot;
        rowinv[t]       = inv;
        rowstat[t]      = P * inv;            // L1 row contribution
        rowstat[16 + t] = qsum * inv * inv;   // L2 row contribution
    }
    __syncthreads();

    if (t == 0) {
        float l1 = 0.f, l2 = 0.f;
        for (int rr = 0; rr < 16; ++rr) { l1 += rowstat[rr]; l2 += rowstat[16 + rr]; }
        l1part[blockIdx.x] = l1;
        l2part[blockIdx.x] = l2;
    }

    // g_in = (S@x) tile = Tfull * rowinv  (reuse Tst)
    float* g_in = Tst;
    for (int q = 0; q < 4; ++q) {
        const int idx = t + 256 * q;
        g_in[idx] = Tfull[idx] * rowinv[idx >> 6];
    }
    __syncthreads();

    // GNN layer: gl = leaky(g_in @ Wg.T + bg)   [16][64]
    for (int q = 0; q < 4; ++q) {
        const int idx = t + 256 * q;
        const int rr = idx >> 6, o = idx & 63;
        float acc = bg[o];
        const float* wrow = Wg + o * 64;
        for (int d = 0; d < 64; ++d) acc += g_in[rr * 64 + d] * wrow[d];
        gl[idx] = leaky(acc);
    }
    __syncthreads();

    // head: out = gl @ W2.T + b2   [16][32]
    for (int q = 0; q < 2; ++q) {
        const int idx = t + 256 * q;
        const int rr = idx >> 5, p = idx & 31;
        float acc = b2[p];
        const float* wrow = W2 + p * 64;
        for (int o = 0; o < 64; ++o) acc += gl[rr * 64 + o] * wrow[o];
        out[(size_t)(i0 + rr) * 32 + p] = acc;
    }
}

// ---------------------------------------------------------------------------
// Kernel 3: deterministic reduction of per-block L1/L2 partials.
// ---------------------------------------------------------------------------
__global__ __launch_bounds__(256) void reduce_kernel(
    const float* __restrict__ l1part, const float* __restrict__ l2part,
    float* __restrict__ out)
{
    __shared__ float r1[256], r2[256];
    const int t = threadIdx.x;
    r1[t] = l1part[t] + l1part[t + 256];
    r2[t] = l2part[t] + l2part[t + 256];
    __syncthreads();
    for (int s = 128; s > 0; s >>= 1) {
        if (t < s) { r1[t] += r1[t + s]; r2[t] += r2[t + s]; }
        __syncthreads();
    }
    if (t == 0) {
        const double invNN = 1.0 / ((double)kN * (double)kN);
        out[(size_t)kN * 32]     = (float)(r1[0] * invNN);
        out[(size_t)kN * 32 + 1] = (float)(r2[0] * invNN);
    }
}

// ---------------------------------------------------------------------------
extern "C" void kernel_launch(void* const* d_in, const int* in_sizes, int n_in,
                              void* d_out, int out_size, void* d_ws, size_t ws_size,
                              hipStream_t stream)
{
    const float* x   = (const float*)d_in[0];
    const float* A   = (const float*)d_in[1];
    const float* W10 = (const float*)d_in[2];
    const float* b10 = (const float*)d_in[3];
    const float* W11 = (const float*)d_in[4];
    const float* b11 = (const float*)d_in[5];
    const float* a   = (const float*)d_in[6];
    const float* Wg  = (const float*)d_in[7];
    const float* bg  = (const float*)d_in[8];
    const float* W2  = (const float*)d_in[9];
    const float* b2  = (const float*)d_in[10];
    float* out = (float*)d_out;

    // workspace carve-out (~3.2 MB total)
    char* ws = (char*)d_ws;
    float*  hf32 = (float*)ws;  ws += (size_t)kN * kDH * sizeof(float);
    __bf16* ht   = (__bf16*)ws; ws += (size_t)kN * kDH * sizeof(__bf16);
    float*  s_ws = (float*)ws;  ws += (size_t)kN * sizeof(float);
    float*  sq_ws= (float*)ws;  ws += (size_t)kN * sizeof(float);
    float*  l1p  = (float*)ws;  ws += 512 * sizeof(float);
    float*  l2p  = (float*)ws;  ws += 512 * sizeof(float);

    mlp_kernel<<<kN / 16, 256, 0, stream>>>(x, W10, b10, W11, b11, a,
                                            hf32, ht, s_ws, sq_ws);
    spass_kernel<<<kN / 16, 256, 0, stream>>>(A, s_ws, sq_ws, ht, hf32,
                                              Wg, bg, W2, b2, out, l1p, l2p);
    reduce_kernel<<<1, 256, 0, stream>>>(l1p, l2p, out);
}
